// SwinBlock_22067541967781
// MI455X (gfx1250) — matrix operand order repaired and checked by reading, not compiled, not run on hardware
//
#include <hip/hip_runtime.h>
#include <math.h>

typedef _Float16 f16;
typedef __attribute__((ext_vector_type(16))) _Float16 v16h;
typedef __attribute__((ext_vector_type(8)))  float    v8f;

#define TPB 256   // 8 wave32
#define C_DIM 192
#define C3 576
#define NHEADS 6
#define HD 32
#define NTOK 64
#define HID 768

// ---------------------------------------------------------------------------
// Fragment loaders (contiguous k order k = 16*(l>>4) + s, same for A and B, wave32)
// ---------------------------------------------------------------------------

// A fragment: 16x32 f16, row-major source with leading dim lda (elements).
// lane l: row m = l&15 ; k = kbase + (l>>4)*16 + 0..15
__device__ __forceinline__ v16h frag_a(const f16* __restrict__ A, int lda, int kbase, int lane) {
  const int m  = lane & 15;
  const int kh = (lane >> 4) << 4;
  const f16* p = A + m * lda + kbase + kh;
  v16h a;
#pragma unroll
  for (int v = 0; v < 8; ++v) a[v] = p[v];
#pragma unroll
  for (int v = 0; v < 8; ++v) a[8 + v] = p[8 + v];
  return a;
}

// B fragment: 32x16 f16 from K-contiguous ("transposed") storage Bt[n][k],
// row stride ldb. lane l: col n = nbase + (l&15); k = kbase + (l>>4)*16 + 0..15
__device__ __forceinline__ v16h frag_bt(const f16* __restrict__ Bt, int ldb, int nbase, int kbase, int lane) {
  const int n  = nbase + (lane & 15);
  const int ks = (lane >> 4) << 4;
  const f16* p = Bt + n * ldb + kbase + ks;
  v16h b;
#pragma unroll
  for (int v = 0; v < 16; ++v) b[v] = p[v];
  return b;
}

// B fragment from row-major B[k][n] (n contiguous), k stride ldb.
__device__ __forceinline__ v16h frag_brm(const f16* __restrict__ B, int ldb, int nbase, int kbase, int lane) {
  const int n  = nbase + (lane & 15);
  const int ks = (lane >> 4) << 4;
  const f16* p = B + (kbase + ks) * ldb + n;
  v16h b;
#pragma unroll
  for (int v = 0; v < 16; ++v) b[v] = p[v * ldb];
  return b;
}

#define WMMA_F16(a, b, c) \
  __builtin_amdgcn_wmma_f32_16x16x32_f16(false, (a), false, (b), (short)0, (c), false, false)

// ---------------------------------------------------------------------------
// Kernel 0: convert weights f32 -> f16, K-contiguous (transposed) layout
//   ws layout: qkvT[576][192] | projT[192][192] | fc1T[768][192] | fc2T[192][768]
// ---------------------------------------------------------------------------
__global__ void prep_weights(const float* __restrict__ qkv_w, const float* __restrict__ proj_w,
                             const float* __restrict__ fc1_w, const float* __restrict__ fc2_w,
                             f16* __restrict__ wsp) {
  f16* qkvT = wsp;
  f16* projT = qkvT + C3 * C_DIM;
  f16* fc1T  = projT + C_DIM * C_DIM;
  f16* fc2T  = fc1T + HID * C_DIM;
  const int i0 = blockIdx.x * blockDim.x + threadIdx.x;
  const int st = gridDim.x * blockDim.x;
  for (int i = i0; i < C3 * C_DIM; i += st)  { int n = i / C_DIM, k = i % C_DIM; qkvT[i] = (f16)qkv_w[k * C3 + n]; }
  for (int i = i0; i < C_DIM * C_DIM; i += st){ int n = i / C_DIM, k = i % C_DIM; projT[i] = (f16)proj_w[k * C_DIM + n]; }
  for (int i = i0; i < HID * C_DIM; i += st) { int n = i / C_DIM, k = i % C_DIM; fc1T[i] = (f16)fc1_w[k * HID + n]; }
  for (int i = i0; i < C_DIM * HID; i += st) { int n = i / HID,  k = i % HID;  fc2T[i] = (f16)fc2_w[k * C_DIM + n]; }
}

// ---------------------------------------------------------------------------
// Kernel 1: fused shift-window attention. 1 block = 1 window (64 tokens).
//   y = x + proj(attn(LN1(window(x)))) scattered back through the roll map.
// ---------------------------------------------------------------------------
// LDS budget (123,136 B):
//   [0,73728)           qkv16 [64][576] f16   (first used as xin [64][192] f32)
//   [73728,98304)       hbuf  [64][192] f16   (LN output, then attn output)
//   [98304,114688)      S     [64][64]  f32
//   [114688,122880)     P     [64][64]  f16
//   [122880,123136)     gidx  [64]      int
__global__ __launch_bounds__(TPB) void swin_attn(
    const float* __restrict__ x, const float* __restrict__ g1, const float* __restrict__ b1,
    const float* __restrict__ qkv_b, const float* __restrict__ proj_b,
    const f16* __restrict__ wsp, float* __restrict__ y) {
  __shared__ __align__(16) char smA[73728 + 24576 + 16384 + 8192 + 256];
  f16*   qkv16 = (f16*)smA;
  float* xin   = (float*)smA;                       // aliases qkv16 (dead before GEMM writes)
  f16*   hbuf  = (f16*)(smA + 73728);
  float* S     = (float*)(smA + 73728 + 24576);
  f16*   P     = (f16*)(smA + 73728 + 24576 + 16384);
  int*   gidx  = (int*)(smA + 73728 + 24576 + 16384 + 8192);

  const f16* qkvT  = wsp;
  const f16* projT = wsp + C3 * C_DIM;

  const int tid  = threadIdx.x;
  const int wave = tid >> 5;
  const int lane = tid & 31;

  // ---- window/shift gather-scatter index (roll(-4) in, roll(+4) out cancel)
  const int b  = blockIdx.x >> 8;
  const int wn = blockIdx.x & 255;
  const int wy = wn >> 4, wx = wn & 15;
  if (tid < NTOK) {
    const int r = tid >> 3, c = tid & 7;
    const int ys = (wy * 8 + r + 4) & 127;
    const int xs = (wx * 8 + c + 4) & 127;
    gidx[tid] = ((b << 14) + (ys << 7) + xs) * C_DIM;
  }
  __syncthreads();

  // ---- stage x window into LDS (f32)
  for (int i = tid; i < NTOK * C_DIM; i += TPB) {
    const int m = i / C_DIM, ch = i % C_DIM;
    xin[i] = x[gidx[m] + ch];
  }
  __syncthreads();

  // ---- LayerNorm1 -> hbuf (f16)
  if (tid < NTOK) {
    const float* row = xin + tid * C_DIM;
    float s = 0.f, s2 = 0.f;
    for (int ch = 0; ch < C_DIM; ++ch) { const float v = row[ch]; s += v; s2 += v * v; }
    const float mu = s * (1.f / C_DIM);
    const float var = s2 * (1.f / C_DIM) - mu * mu;
    const float rs = rsqrtf(var + 1e-5f);
    for (int ch = 0; ch < C_DIM; ++ch)
      hbuf[tid * C_DIM + ch] = (f16)(((row[ch] - mu) * rs) * g1[ch] + b1[ch]);
  }
  __syncthreads();

  // ---- qkv GEMM: [64,192] x [192,576] -> qkv16 (overwrites dead xin)
  for (int t = wave; t < 4 * 36; t += 8) {
    const int mt = t / 36, nt = t % 36;
    v8f acc = {};
#pragma unroll
    for (int kt = 0; kt < 6; ++kt) {
      const v16h a  = frag_a(hbuf + mt * 16 * C_DIM, C_DIM, kt * 32, lane);
      const v16h bb = frag_bt(qkvT, C_DIM, nt * 16, kt * 32, lane);
      acc = WMMA_F16(a, bb, acc);
    }
    const int n = nt * 16 + (lane & 15);
    const float bias = qkv_b[n];
#pragma unroll
    for (int r = 0; r < 8; ++r) {
      const int m = mt * 16 + r + ((lane >> 4) << 3);
      qkv16[m * C3 + n] = (f16)(acc[r] + bias);
    }
  }
  __syncthreads();

  // ---- attention per head: S = Q K^T * scale ; softmax ; O = P V
  const float scale = 0.17677669529663687f;  // 32^-0.5
  for (int h = 0; h < NHEADS; ++h) {
    // S tiles (4x4 of 16x16), K = 32 (one WMMA)
    for (int t = wave; t < 16; t += 8) {
      const int mt = t >> 2, nt = t & 3;
      const v16h a  = frag_a(qkv16 + h * HD + mt * 16 * C3, C3, 0, lane);
      const v16h bb = frag_bt(qkv16 + C_DIM + h * HD, C3, nt * 16, 0, lane);  // K^T is k-contiguous
      v8f acc = {};
      acc = WMMA_F16(a, bb, acc);
      const int n = nt * 16 + (lane & 15);
#pragma unroll
      for (int r = 0; r < 8; ++r) {
        const int m = mt * 16 + r + ((lane >> 4) << 3);
        S[m * NTOK + n] = acc[r] * scale;
      }
    }
    __syncthreads();
    // softmax rows -> P (f16)
    if (tid < NTOK) {
      float* srow = S + tid * NTOK;
      float mx = srow[0];
      for (int j = 1; j < NTOK; ++j) mx = fmaxf(mx, srow[j]);
      float sum = 0.f;
      for (int j = 0; j < NTOK; ++j) { const float e = expf(srow[j] - mx); srow[j] = e; sum += e; }
      const float inv = 1.f / sum;
      for (int j = 0; j < NTOK; ++j) P[tid * NTOK + j] = (f16)(srow[j] * inv);
    }
    __syncthreads();
    // O = P V : M=64, N=32, K=64  (8 tiles, 1 per wave)
    {
      const int t = wave;  // 0..7
      const int mt = t >> 1, nt = t & 1;
      v8f acc = {};
#pragma unroll
      for (int kt = 0; kt < 2; ++kt) {
        const v16h a  = frag_a(P + mt * 16 * NTOK, NTOK, kt * 32, lane);
        const v16h bb = frag_brm(qkv16 + 2 * C_DIM + h * HD, C3, nt * 16, kt * 32, lane);  // V row-major
        acc = WMMA_F16(a, bb, acc);
      }
      const int n = nt * 16 + (lane & 15);
#pragma unroll
      for (int r = 0; r < 8; ++r) {
        const int m = mt * 16 + r + ((lane >> 4) << 3);
        hbuf[m * C_DIM + h * HD + n] = (f16)acc[r];  // hbuf reused as attn-out
      }
    }
    __syncthreads();
  }

  // ---- proj GEMM + bias + residual, scatter back through roll map
  for (int t = wave; t < 4 * 12; t += 8) {
    const int mt = t / 12, nt = t % 12;
    v8f acc = {};
#pragma unroll
    for (int kt = 0; kt < 6; ++kt) {
      const v16h a  = frag_a(hbuf + mt * 16 * C_DIM, C_DIM, kt * 32, lane);
      const v16h bb = frag_bt(projT, C_DIM, nt * 16, kt * 32, lane);
      acc = WMMA_F16(a, bb, acc);
    }
    const int n = nt * 16 + (lane & 15);
    const float pb = proj_b[n];
#pragma unroll
    for (int r = 0; r < 8; ++r) {
      const int m = mt * 16 + r + ((lane >> 4) << 3);
      const int g = gidx[m] + n;
      y[g] = x[g] + acc[r] + pb;
    }
  }
}

// ---------------------------------------------------------------------------
// Kernel 2: fused LN2 + MLP (GELU) + residual, in-place on y (=d_out).
//   1 block = 64 token rows.
// LDS (122,880 B):
//   [0,98304)        hid16 [64][768] f16  (first used as yln [64][192] f32)
//   [98304,122880)   h2    [64][192] f16
// ---------------------------------------------------------------------------
__global__ __launch_bounds__(TPB) void swin_mlp(
    const float* __restrict__ g2, const float* __restrict__ b2,
    const float* __restrict__ fc1_b, const float* __restrict__ fc2_b,
    const f16* __restrict__ wsp, float* __restrict__ y) {
  __shared__ __align__(16) char smB[98304 + 24576];
  f16*   hid = (f16*)smB;
  float* yln = (float*)smB;  // alias, dead before hid is written
  f16*   h2  = (f16*)(smB + 98304);

  const f16* fc1T = wsp + C3 * C_DIM + C_DIM * C_DIM;
  const f16* fc2T = fc1T + HID * C_DIM;

  const int tid  = threadIdx.x;
  const int wave = tid >> 5;
  const int lane = tid & 31;
  const long row0 = (long)blockIdx.x * NTOK;

  // stage y rows
  for (int i = tid; i < NTOK * C_DIM; i += TPB) yln[i] = y[row0 * C_DIM + i];
  __syncthreads();

  // LN2 -> h2 (f16)
  if (tid < NTOK) {
    const float* row = yln + tid * C_DIM;
    float s = 0.f, s2 = 0.f;
    for (int ch = 0; ch < C_DIM; ++ch) { const float v = row[ch]; s += v; s2 += v * v; }
    const float mu = s * (1.f / C_DIM);
    const float var = s2 * (1.f / C_DIM) - mu * mu;
    const float rs = rsqrtf(var + 1e-5f);
    for (int ch = 0; ch < C_DIM; ++ch)
      h2[tid * C_DIM + ch] = (f16)(((row[ch] - mu) * rs) * g2[ch] + b2[ch]);
  }
  __syncthreads();

  // fc1 + exact GELU -> hid (overwrites dead yln)
  for (int t = wave; t < 4 * 48; t += 8) {
    const int mt = t / 48, nt = t % 48;
    v8f acc = {};
#pragma unroll
    for (int kt = 0; kt < 6; ++kt) {
      const v16h a  = frag_a(h2 + mt * 16 * C_DIM, C_DIM, kt * 32, lane);
      const v16h bb = frag_bt(fc1T, C_DIM, nt * 16, kt * 32, lane);
      acc = WMMA_F16(a, bb, acc);
    }
    const int n = nt * 16 + (lane & 15);
    const float bias = fc1_b[n];
#pragma unroll
    for (int r = 0; r < 8; ++r) {
      const int m = mt * 16 + r + ((lane >> 4) << 3);
      const float v = acc[r] + bias;
      const float ge = 0.5f * v * (1.f + erff(v * 0.7071067811865476f));
      hid[m * HID + n] = (f16)ge;
    }
  }
  __syncthreads();

  // fc2 + bias + residual (read-modify-write of y; each element by one lane)
  for (int t = wave; t < 4 * 12; t += 8) {
    const int mt = t / 12, nt = t % 12;
    v8f acc = {};
#pragma unroll
    for (int kt = 0; kt < 24; ++kt) {
      const v16h a  = frag_a(hid + mt * 16 * HID, HID, kt * 32, lane);
      const v16h bb = frag_bt(fc2T, HID, nt * 16, kt * 32, lane);
      acc = WMMA_F16(a, bb, acc);
    }
    const int n = nt * 16 + (lane & 15);
    const float bias = fc2_b[n];
#pragma unroll
    for (int r = 0; r < 8; ++r) {
      const int m = mt * 16 + r + ((lane >> 4) << 3);
      const long g = (row0 + m) * C_DIM + n;
      y[g] = y[g] + acc[r] + bias;
    }
  }
}

// ---------------------------------------------------------------------------
extern "C" void kernel_launch(void* const* d_in, const int* in_sizes, int n_in,
                              void* d_out, int out_size, void* d_ws, size_t ws_size,
                              hipStream_t stream) {
  const float* x      = (const float*)d_in[0];
  const float* gamma1 = (const float*)d_in[1];
  const float* beta1  = (const float*)d_in[2];
  const float* qkv_w  = (const float*)d_in[3];
  const float* qkv_b  = (const float*)d_in[4];
  const float* proj_w = (const float*)d_in[5];
  const float* proj_b = (const float*)d_in[6];
  const float* gamma2 = (const float*)d_in[7];
  const float* beta2  = (const float*)d_in[8];
  const float* fc1_w  = (const float*)d_in[9];
  const float* fc1_b  = (const float*)d_in[10];
  const float* fc2_w  = (const float*)d_in[11];
  const float* fc2_b  = (const float*)d_in[12];
  // d_in[13], d_in[14] are H, W (compile-time constants here: 128, 128)

  f16*   wsp = (f16*)d_ws;
  float* out = (float*)d_out;

  prep_weights<<<512, TPB, 0, stream>>>(qkv_w, proj_w, fc1_w, fc2_w, wsp);
  swin_attn<<<4096, TPB, 0, stream>>>(x, gamma1, beta1, qkv_b, proj_b, wsp, out);
  swin_mlp<<<4096, TPB, 0, stream>>>(gamma2, beta2, fc1_b, fc2_b, wsp, out);
}
